// TwoWordPSDProbe_34505767256461
// MI455X (gfx1250) — matrix operand order repaired and checked by reading, not compiled, not run on hardware
//
#include <hip/hip_runtime.h>
#include <hip/hip_bf16.h>

// ---------------------------------------------------------------------------
// MI455X (gfx1250): bf16 V_WMMA_F32_16X16X32_BF16 for all matmul stages,
// fp32 accumulation. GEMMs are LDS-staged via the CDNA5 async-copy path
// (global_load_async_to_lds_b128 + s_wait_asynccnt), double buffered so the
// async copies of K-step s+1 overlap the 8 WMMAs of step s.
// wave32, 8 waves/block, 128x128 block tile, 64x32 wave tile.
// ---------------------------------------------------------------------------

typedef __bf16 bf16_t;
typedef bf16_t v16bf __attribute__((ext_vector_type(16)));
typedef float  v8f   __attribute__((ext_vector_type(8)));

union Frag16 { v16bf v; unsigned u[8]; };

__device__ __forceinline__ unsigned short f32_to_bf16_raw(float f) {
    unsigned u = __float_as_uint(f);
    u += 0x7FFFu + ((u >> 16) & 1u);   // round-to-nearest-even
    return (unsigned short)(u >> 16);
}
__device__ __forceinline__ float bf16_raw_to_f32(unsigned short h) {
    return __uint_as_float(((unsigned)h) << 16);
}

// A fragment: 16x32 bf16 tile (row-major source, row stride = ld ushorts).
// ISA layout: lanes 0-15 = M, half=lane>>4 selects K interleave:
//   VGPR 0..3 : K = 8*half + 2i, 2i+1 ; VGPR 4..7 : K = 16 + 8*half + 2i
__device__ __forceinline__ v16bf load_a_frag(const unsigned short* A,
                                             size_t row, int ld, int k0, int lane) {
    Frag16 f;
    const unsigned short* p = A + row * (size_t)ld + k0;
    const int h8 = (lane >> 4) << 3;
#pragma unroll
    for (int i = 0; i < 4; ++i)
        f.u[i] = *(const unsigned*)(p + h8 + 2 * i);
#pragma unroll
    for (int i = 0; i < 4; ++i)
        f.u[4 + i] = *(const unsigned*)(p + 16 + h8 + 2 * i);
    return f.v;
}

// B fragment: 32x16 bf16 tile, source stored N-major (row n holds K contig).
// Same K interleave as A: VGPR 0..3 : K = 8*half + 2i ; VGPR 4..7 : K = 16 + 8*half + 2i
__device__ __forceinline__ v16bf load_b_frag(const unsigned short* Bt,
                                             size_t row, int ld, int k0, int lane) {
    Frag16 f;
    const unsigned short* p = Bt + row * (size_t)ld + k0 + ((lane >> 4) << 3);
#pragma unroll
    for (int i = 0; i < 8; ++i)
        f.u[i] = *(const unsigned*)(p + 2 * i + ((i >> 2) << 3));
    return f.v;
}

// LDS row stride for staged tiles: 32 K-elements + 8 pad ushorts (80 B row,
// 16 B aligned for b128 async stores, 16-bank-clean fragment reads).
#define ALD 40

// ---------------------------------------------------------------------------
// C = relu(A[M,K] x Bt[N,K]^T + bias[N]) -> bf16 C[M,N]
// grid: (N/128, M/128), block: 256 threads (8 waves, 2(M) x 4(N) wave grid)
// ---------------------------------------------------------------------------
__global__ __launch_bounds__(256)
void gemm_bf16_relu_kernel(const unsigned short* __restrict__ A,
                           const unsigned short* __restrict__ Bt,
                           const float* __restrict__ bias,
                           unsigned short* __restrict__ C,
                           int M, int N, int K) {
    __shared__ unsigned short Abuf[2][128 * ALD];
    __shared__ unsigned short Bbuf[2][128 * ALD];

    const int tid     = threadIdx.x;
    const int lane    = tid & 31;
    const int wave    = tid >> 5;
    const int m_block = blockIdx.y * 128;
    const int n_block = blockIdx.x * 128;
    const int m_wave  = (wave >> 2) * 64;   // 4 M-tiles of 16
    const int n_wave  = (wave & 3) * 32;    // 2 N-tiles of 16
    const int l15     = lane & 15;

    // Async stage: copy A[128 x 32] and Bt[128 x 32] slabs into LDS.
    // 512 x 16B per slab; 256 threads -> 2 b128 ops per thread per slab.
    auto issue_stage = [&](int buf, int k0) {
#pragma unroll
        for (int rep = 0; rep < 2; ++rep) {
            const int linear = rep * 256 + tid;     // [0,512)
            const int row = linear >> 2;
            const int seg = (linear & 3) * 8;       // ushort offset of 16B seg
            {
                const unsigned short* g = A + (size_t)(m_block + row) * K + k0 + seg;
                unsigned l = (unsigned)(size_t)&Abuf[buf][row * ALD + seg];
                asm volatile("global_load_async_to_lds_b128 %0, %1, off"
                             :: "v"(l), "v"((unsigned long long)(size_t)g) : "memory");
            }
            {
                const unsigned short* g = Bt + (size_t)(n_block + row) * K + k0 + seg;
                unsigned l = (unsigned)(size_t)&Bbuf[buf][row * ALD + seg];
                asm volatile("global_load_async_to_lds_b128 %0, %1, off"
                             :: "v"(l), "v"((unsigned long long)(size_t)g) : "memory");
            }
        }
    };

    v8f acc[4][2];
    const v8f vzero = {};
#pragma unroll
    for (int mt = 0; mt < 4; ++mt)
#pragma unroll
        for (int nt = 0; nt < 2; ++nt) acc[mt][nt] = vzero;

    const int nsteps = K >> 5;
    issue_stage(0, 0);

    for (int s = 0; s < nsteps; ++s) {
        const int buf = s & 1;
        // my async writes for stage s are done ...
        asm volatile("s_wait_asynccnt 0x0" ::: "memory");
        // ... and everyone else's too; also all reads of buf^1 from stage s-1
        // are complete, so it is safe to overwrite buf^1 below.
        __syncthreads();
        // prefetch stage s+1 (clamped on the last step: dead loads, never read;
        // S_ENDPGM's implicit wait-idle drains them)
        const int knext = (s + 1 < nsteps) ? (s + 1) << 5 : s << 5;
        issue_stage(buf ^ 1, knext);

        v16bf af[4], bfr[2];
#pragma unroll
        for (int mt = 0; mt < 4; ++mt)
            af[mt] = load_a_frag(&Abuf[buf][0], (size_t)(m_wave + mt * 16 + l15), ALD, 0, lane);
#pragma unroll
        for (int nt = 0; nt < 2; ++nt)
            bfr[nt] = load_b_frag(&Bbuf[buf][0], (size_t)(n_wave + nt * 16 + l15), ALD, 0, lane);

#pragma unroll
        for (int mt = 0; mt < 4; ++mt)
#pragma unroll
            for (int nt = 0; nt < 2; ++nt)
                acc[mt][nt] = __builtin_amdgcn_wmma_f32_16x16x32_bf16(
                    false, af[mt], false, bfr[nt], (short)0, acc[mt][nt], false, false);
    }

    // D layout: n = lane&15, m = vgpr + 8*(lane>>4)
    const int half = lane >> 4;
#pragma unroll
    for (int nt = 0; nt < 2; ++nt) {
        const int n = n_block + n_wave + nt * 16 + l15;
        const float bv = bias[n];
#pragma unroll
        for (int mt = 0; mt < 4; ++mt) {
#pragma unroll
            for (int r = 0; r < 8; ++r) {
                const int m = m_block + m_wave + mt * 16 + r + 8 * half;
                float v = acc[mt][nt][r] + bv;
                v = v > 0.0f ? v : 0.0f;
                C[(size_t)m * N + n] = f32_to_bf16_raw(v);
            }
        }
    }
}

// ---------------------------------------------------------------------------
// Per-batch pairwise squared distances:
//   out[b,i,j] = max(sq[b,i] + sq[b,j] - 2 * (T_b T_b^T)[i,j], 0)
// T: bf16 [B*L, R] row-major; grid (L/128, L/128, B); block 256.
// K = R = 128 only; rows stream straight from L2 (T is 4 MB, fully resident).
// ---------------------------------------------------------------------------
__global__ __launch_bounds__(256)
void gram_dist_kernel(const unsigned short* __restrict__ T,
                      const float* __restrict__ sq,
                      float* __restrict__ out, int L, int R) {
    const int b = blockIdx.z;
    const unsigned short* Tb  = T  + (size_t)b * L * R;
    const float*          sqb = sq + (size_t)b * L;

    const int lane   = threadIdx.x & 31;
    const int wave   = threadIdx.x >> 5;
    const int i_wave = blockIdx.y * 128 + (wave >> 2) * 64;
    const int j_wave = blockIdx.x * 128 + (wave & 3) * 32;
    const int l15    = lane & 15;

    v8f acc[4][2];
    const v8f vzero = {};
#pragma unroll
    for (int mt = 0; mt < 4; ++mt)
#pragma unroll
        for (int nt = 0; nt < 2; ++nt) acc[mt][nt] = vzero;

#pragma unroll
    for (int k0 = 0; k0 < 128; k0 += 32) {   // R == 128, 4 steps
        v16bf af[4], bfr[2];
#pragma unroll
        for (int mt = 0; mt < 4; ++mt)
            af[mt] = load_a_frag(Tb, (size_t)(i_wave + mt * 16 + l15), R, k0, lane);
#pragma unroll
        for (int nt = 0; nt < 2; ++nt)
            bfr[nt] = load_b_frag(Tb, (size_t)(j_wave + nt * 16 + l15), R, k0, lane);
#pragma unroll
        for (int mt = 0; mt < 4; ++mt)
#pragma unroll
            for (int nt = 0; nt < 2; ++nt)
                acc[mt][nt] = __builtin_amdgcn_wmma_f32_16x16x32_bf16(
                    false, af[mt], false, bfr[nt], (short)0, acc[mt][nt], false, false);
    }

    const int half = lane >> 4;
    float* outb = out + (size_t)b * L * L;
#pragma unroll
    for (int nt = 0; nt < 2; ++nt) {
        const int j = j_wave + nt * 16 + l15;
        const float sj = sqb[j];
#pragma unroll
        for (int mt = 0; mt < 4; ++mt) {
#pragma unroll
            for (int r = 0; r < 8; ++r) {
                const int i = i_wave + mt * 16 + r + 8 * half;
                float d = sqb[i] + sj - 2.0f * acc[mt][nt][r];
                outb[(size_t)i * L + j] = d > 0.0f ? d : 0.0f;
            }
        }
    }
}

// ---------------------------------------------------------------------------
// Helper elementwise kernels
// ---------------------------------------------------------------------------
__global__ void cvt_f32_bf16_kernel(const float* __restrict__ in,
                                    unsigned short* __restrict__ out, int n) {
    int i = blockIdx.x * blockDim.x + threadIdx.x;
    if (i < n) out[i] = f32_to_bf16_raw(in[i]);
}

// out[r, c] (rows_out x cols_out) = cvt(in[c, r]), in is [cols_out, rows_out]
__global__ void transpose_cvt_kernel(const float* __restrict__ in,
                                     unsigned short* __restrict__ out,
                                     int rows_out, int cols_out) {
    int i = blockIdx.x * blockDim.x + threadIdx.x;
    if (i < rows_out * cols_out) {
        int r = i / cols_out;
        int c = i - r * cols_out;
        out[i] = f32_to_bf16_raw(in[(size_t)c * rows_out + r]);
    }
}

__global__ void sqnorm_kernel(const unsigned short* __restrict__ T,
                              float* __restrict__ sq, int rows, int R) {
    int i = blockIdx.x * blockDim.x + threadIdx.x;
    if (i < rows) {
        const unsigned short* p = T + (size_t)i * R;
        float s = 0.0f;
        for (int r = 0; r < R; ++r) {
            float v = bf16_raw_to_f32(p[r]);
            s += v * v;
        }
        sq[i] = s;
    }
}

// ---------------------------------------------------------------------------
// Launch
// ---------------------------------------------------------------------------
extern "C" void kernel_launch(void* const* d_in, const int* in_sizes, int n_in,
                              void* d_out, int out_size, void* d_ws, size_t ws_size,
                              hipStream_t stream) {
    (void)in_sizes; (void)n_in; (void)out_size; (void)ws_size;

    constexpr int B = 16, L = 1024, D = 1024, H = 1024, R = 128;
    constexpr int M = B * L;                 // 16384

    const float* batch = (const float*)d_in[0];   // [B,L,D]
    const float* W1    = (const float*)d_in[1];   // [D,H]
    const float* b1    = (const float*)d_in[2];   // [H]
    const float* W2    = (const float*)d_in[3];   // [H,R]
    const float* b2    = (const float*)d_in[4];   // [R]
    float* out = (float*)d_out;                   // [B,L,L]

    // workspace layout (bf16 stored as ushort raw)
    unsigned short* Xbf = (unsigned short*)d_ws;              // M*D
    unsigned short* Hbf = Xbf + (size_t)M * D;                // M*H
    unsigned short* W1t = Hbf + (size_t)M * H;                // H*D  (N-major)
    unsigned short* W2t = W1t + (size_t)H * D;                // R*H  (N-major)
    unsigned short* Tbf = W2t + (size_t)R * H;                // M*R
    float*          sq  = (float*)(Tbf + (size_t)M * R);      // M

    // 1) convert inputs to bf16 (B operands transposed to N-major)
    {
        int n = M * D;
        cvt_f32_bf16_kernel<<<(n + 255) / 256, 256, 0, stream>>>(batch, Xbf, n);
        transpose_cvt_kernel<<<(H * D + 255) / 256, 256, 0, stream>>>(W1, W1t, H, D);
        transpose_cvt_kernel<<<(R * H + 255) / 256, 256, 0, stream>>>(W2, W2t, R, H);
    }

    // 2) H = relu(X W1 + b1)   [16384 x 1024] x [1024 x 1024]
    gemm_bf16_relu_kernel<<<dim3(H / 128, M / 128), 256, 0, stream>>>(
        Xbf, W1t, b1, Hbf, M, H, D);

    // 3) T = relu(H W2 + b2)   [16384 x 1024] x [1024 x 128]
    gemm_bf16_relu_kernel<<<dim3(R / 128, M / 128), 256, 0, stream>>>(
        Hbf, W2t, b2, Tbf, M, R, H);

    // 4) row squared norms of T
    sqnorm_kernel<<<(M + 255) / 256, 256, 0, stream>>>(Tbf, sq, M, R);

    // 5) out[b,i,j] = max(sq_i + sq_j - 2 T_b T_b^T, 0)
    gram_dist_kernel<<<dim3(L / 128, L / 128, B), 256, 0, stream>>>(
        Tbf, sq, out, L, R);
}
